// DotGATLayer_42064909697461
// MI455X (gfx1250) — matrix operand order repaired and checked by reading, not compiled, not run on hardware
//
#include <hip/hip_runtime.h>

#define B_ 8
#define A_ 2048
#define H_ 128
#define D_ 128
#define INV_SCALE 0.08838834764831843f   /* 1/sqrt(128) */
#define LN_EPS 1e-5f

typedef __attribute__((ext_vector_type(16))) _Float16 v16h;
typedef __attribute__((ext_vector_type(8)))  _Float16 v8h;
typedef __attribute__((ext_vector_type(8)))  float    v8f;
typedef __attribute__((ext_vector_type(4)))  float    v4f;

// ---- WMMA fragment loaders (layouts per CDNA5 ISA §7.12.2) ----
// A (16x32 f16): lane m=L&15, half=L>>4; halves [0..7] = K in [8*half, 8*half+8),
//                halves [8..15] = K in [16+8*half, 16+8*half+8).
__device__ __forceinline__ v16h load_a_f16(const _Float16* rowp, int kb, int half) {
  v8h lo = *(const v8h*)(rowp + kb + 8 * half);
  v8h hi = *(const v8h*)(rowp + kb + 16 + 8 * half);
  v16h a;
#pragma unroll
  for (int e = 0; e < 8; ++e) { a[e] = lo[e]; a[8 + e] = hi[e]; }
  return a;
}

__device__ __forceinline__ v16h load_a_f32cvt(const float* rowp, int kb, int half) {
  v4f f0 = *(const v4f*)(rowp + kb + 8 * half);
  v4f f1 = *(const v4f*)(rowp + kb + 8 * half + 4);
  v4f f2 = *(const v4f*)(rowp + kb + 16 + 8 * half);
  v4f f3 = *(const v4f*)(rowp + kb + 16 + 8 * half + 4);
  v16h a;
#pragma unroll
  for (int e = 0; e < 4; ++e) {
    a[e]      = (_Float16)f0[e];
    a[4 + e]  = (_Float16)f1[e];
    a[8 + e]  = (_Float16)f2[e];
    a[12 + e] = (_Float16)f3[e];
  }
  return a;
}

// B (32x16 f16): lane n=L&15, half=L>>4; same K order as A (8*half + e, 16+8*half + e-8).
__device__ __forceinline__ v16h load_b_f16(const _Float16* rowp, int kb, int half) {
  v8h lo = *(const v8h*)(rowp + kb + 8 * half);
  v8h hi = *(const v8h*)(rowp + kb + 16 + 8 * half);
  v16h b;
#pragma unroll
  for (int e = 0; e < 8; ++e) { b[e] = lo[e]; b[8 + e] = hi[e]; }
  return b;
}

// ---- Kernel 0: convert+transpose weights to f16 [D][H] ----
__global__ void wconv(const float* __restrict__ Wq, const float* __restrict__ Wk,
                      const float* __restrict__ Wv, _Float16* __restrict__ Wt) {
  int t = blockIdx.x * blockDim.x + threadIdx.x;
  if (t >= 3 * H_ * D_) return;
  int mat = t / (H_ * D_);
  int rem = t - mat * (H_ * D_);
  int d = rem / H_;
  int h = rem - d * H_;
  const float* W = (mat == 0) ? Wq : ((mat == 1) ? Wk : Wv);
  Wt[t] = (_Float16)W[h * D_ + d];   // Wt[mat][d][h] = W[h][d]
}

// ---- Kernel 1: Q/K/V projections via f16 WMMA, f32 accumulate ----
__global__ void __launch_bounds__(256) proj_kernel(
    const float* __restrict__ x, const _Float16* __restrict__ Wt,
    _Float16* __restrict__ Qh, _Float16* __restrict__ Kh, float* __restrict__ V) {
  const int lane = threadIdx.x & 31;
  const int wv   = threadIdx.x >> 5;     // 0..7 -> N tile
  const int r0   = blockIdx.x * 16;      // row tile in [0, B*A)
  const int m    = lane & 15;
  const int half = lane >> 4;
  const int n0   = wv * 16;

  // A fragments for 16 rows of x, converted to f16 on the fly
  v16h ax[4];
  const float* xrow = x + (size_t)(r0 + m) * H_;
#pragma unroll
  for (int kb = 0; kb < 4; ++kb) ax[kb] = load_a_f32cvt(xrow, kb * 32, half);

#pragma unroll
  for (int mat = 0; mat < 3; ++mat) {
    const _Float16* wrow = Wt + mat * (H_ * D_) + (size_t)(n0 + m) * H_;
    v8f acc = {};
#pragma unroll
    for (int kb = 0; kb < 4; ++kb) {
      v16h b = load_b_f16(wrow, kb * 32, half);
      acc = __builtin_amdgcn_wmma_f32_16x16x32_f16(false, ax[kb], false, b,
                                                   (short)0, acc, false, false);
    }
#pragma unroll
    for (int v = 0; v < 8; ++v) {
      int row = r0 + v + 8 * half;     // C layout: M = v + 8*(lane>=16)
      int col = n0 + m;                // N = lane&15
      if (mat == 0)      Qh[(size_t)row * D_ + col] = (_Float16)acc[v];
      else if (mat == 1) Kh[(size_t)row * D_ + col] = (_Float16)acc[v];
      else               V[(size_t)row * D_ + col]  = acc[v];
    }
  }
}

// ---- Kernel 2: fused scores + top-8 + softmax + V-gather + LayerNorm ----
__global__ void __launch_bounds__(128) attn_kernel(
    const _Float16* __restrict__ Qh, const _Float16* __restrict__ Kh,
    const float* __restrict__ V, const float* __restrict__ conn,
    const float* __restrict__ gamma, const float* __restrict__ beta,
    float* __restrict__ out) {
  __shared__ float stage[4][16][20];   // stride 20 floats -> 16B-aligned 8-col runs
  __shared__ float mergev[16][65];
  __shared__ int   mergei[16][65];
  __shared__ float probs[16][8];
  __shared__ int   pidx[16][8];
  __shared__ float otile[16][132];
  __shared__ float muL[16], rsL[16];

  const int lane = threadIdx.x & 31;
  const int wv   = threadIdx.x >> 5;   // 0..3
  const int b    = blockIdx.y;
  const int r0   = blockIdx.x * 16;
  const size_t rowbase = (size_t)b * A_;

  const int n    = lane & 15;
  const int half = lane >> 4;
  const int rr   = lane >> 1;          // row this lane scans for top-k
  const int p    = lane & 1;           // column-half parity

  // Q A-fragments for this 16-row tile (held in registers all loop)
  v16h aq[4];
  {
    const _Float16* qrow = Qh + (rowbase + r0 + n) * (size_t)D_;
#pragma unroll
    for (int kb = 0; kb < 4; ++kb) aq[kb] = load_a_f16(qrow, kb * 32, half);
  }

  float tv[8]; int ti[8];
#pragma unroll
  for (int k = 0; k < 8; ++k) { tv[k] = -__builtin_inff(); ti[k] = 0; }

  const float* connb = conn + (size_t)b * A_ * A_;
  const float* connrow = connb + (size_t)(r0 + rr) * A_ + 8 * p;  // + c0 later

  // No inter-wave data sharing inside this loop: each wave owns stage[wv].
  // Wave-level scheduling barriers are enough (DS ops are wave-wide, in-order),
  // so the 4 waves slip freely and hide the HBM latency of the conn stream.
  for (int it = 0; it < 32; ++it) {
    const int ct = wv + it * 4;
    const int c0 = ct * 16;
    const _Float16* krow = Kh + (rowbase + c0 + n) * (size_t)D_;
    if (it + 1 < 32) {
      __builtin_prefetch((const void*)(krow + 64 * (size_t)D_), 0, 1);
      __builtin_prefetch((const void*)(connrow + c0 + 64), 0, 1);
    }
    v8f acc = {};
#pragma unroll
    for (int kb = 0; kb < 4; ++kb) {
      v16h bfr = load_b_f16(krow, kb * 32, half);   // B = K^T, contiguous features
      acc = __builtin_amdgcn_wmma_f32_16x16x32_f16(false, aq[kb], false, bfr,
                                                   (short)0, acc, false, false);
    }
    // stage scaled scores (C layout scatter: M = v + 8*half, N = n)
#pragma unroll
    for (int v = 0; v < 8; ++v) stage[wv][v + 8 * half][n] = acc[v] * INV_SCALE;
    __builtin_amdgcn_wave_barrier();

    // row-major read-back: 2 lanes/row, 8 contiguous cols each (2x ds_load_b128)
    const float* sp = &stage[wv][rr][8 * p];
    v4f s0v = *(const v4f*)(sp);
    v4f s1v = *(const v4f*)(sp + 4);
    __builtin_amdgcn_wave_barrier();  // keep next iter's stores below these loads

    // coalesced connectivity: 2x global_load_b128 per lane, each elem read once
    const float* cp = connrow + c0;
    v4f c0v = *(const v4f*)(cp);
    v4f c1v = *(const v4f*)(cp + 4);

    float cand[8];
#pragma unroll
    for (int u = 0; u < 4; ++u) {
      cand[u]     = s0v[u] + c0v[u];
      cand[4 + u] = s1v[u] + c1v[u];
    }
    // per-lane running top-8 (threshold test, then unrolled bubble insert)
#pragma unroll
    for (int j = 0; j < 8; ++j) {
      float sv = cand[j];
      int si = c0 + 8 * p + j;
      if (sv > tv[7]) {
        tv[7] = sv; ti[7] = si;
#pragma unroll
        for (int t = 7; t > 0; --t) {
          if (tv[t] > tv[t - 1]) {
            float a0 = tv[t]; tv[t] = tv[t - 1]; tv[t - 1] = a0;
            int i0 = ti[t]; ti[t] = ti[t - 1]; ti[t - 1] = i0;
          }
        }
      }
    }
  }

  // dump 8 lane-local top-8 lists per row (4 waves * 2 lanes)
  {
    int base = (wv * 2 + p) * 8;
#pragma unroll
    for (int k = 0; k < 8; ++k) {
      mergev[rr][base + k] = tv[k];
      mergei[rr][base + k] = ti[k];
    }
  }
  __syncthreads();

  // exact row top-8 of the 64 candidates, then softmax
  if (threadIdx.x < 16) {
    int r = threadIdx.x;
    float vals[8]; int inds[8];
#pragma unroll
    for (int k = 0; k < 8; ++k) {
      float best = -__builtin_inff(); int bj = 0;
      for (int j = 0; j < 64; ++j) {
        float vvv = mergev[r][j];
        if (vvv > best) { best = vvv; bj = j; }
      }
      vals[k] = best; inds[k] = mergei[r][bj];
      mergev[r][bj] = -__builtin_inff();
    }
    float mx = vals[0];
    float ex[8]; float sum = 0.f;
#pragma unroll
    for (int k = 0; k < 8; ++k) { ex[k] = __expf(vals[k] - mx); sum += ex[k]; }
    float inv = 1.f / sum;
#pragma unroll
    for (int k = 0; k < 8; ++k) { probs[r][k] = ex[k] * inv; pidx[r][k] = inds[k]; }
  }
  __syncthreads();

  // out_row = sum_k probs[k] * V[idx[k], :]   (8 threads per row, 16 cols each)
  {
    int r = threadIdx.x >> 3;
    int d0 = (threadIdx.x & 7) * 16;
    float accd[16];
#pragma unroll
    for (int i = 0; i < 16; ++i) accd[i] = 0.f;
#pragma unroll
    for (int k = 0; k < 8; ++k) {
      float pk = probs[r][k];
      const float* vr = V + (rowbase + pidx[r][k]) * (size_t)D_ + d0;
#pragma unroll
      for (int i = 0; i < 16; i += 4) {
        v4f vvv = *(const v4f*)(vr + i);
#pragma unroll
        for (int u = 0; u < 4; ++u) accd[i + u] += pk * vvv[u];
      }
    }
#pragma unroll
    for (int i = 0; i < 16; ++i) otile[r][d0 + i] = accd[i];
  }
  __syncthreads();

  // LayerNorm stats per row
  if (threadIdx.x < 16) {
    int r = threadIdx.x;
    float s = 0.f;
    for (int d = 0; d < D_; ++d) s += otile[r][d];
    float mu = s * (1.0f / D_);
    float vs = 0.f;
    for (int d = 0; d < D_; ++d) { float t = otile[r][d] - mu; vs += t * t; }
    muL[r] = mu;
    rsL[r] = rsqrtf(vs * (1.0f / D_) + LN_EPS);
  }
  __syncthreads();

  {
    int r = threadIdx.x >> 3;
    int d0 = (threadIdx.x & 7) * 16;
    float mu = muL[r], rs = rsL[r];
    float* op = out + (rowbase + r0 + r) * (size_t)D_ + d0;
#pragma unroll
    for (int i = 0; i < 16; ++i) {
      int d = d0 + i;
      op[i] = (otile[r][d] - mu) * rs * gamma[d] + beta[d];
    }
  }
}

extern "C" void kernel_launch(void* const* d_in, const int* in_sizes, int n_in,
                              void* d_out, int out_size, void* d_ws, size_t ws_size,
                              hipStream_t stream) {
  const float* x     = (const float*)d_in[0];
  const float* conn  = (const float*)d_in[1];
  const float* Wq    = (const float*)d_in[2];
  const float* Wk    = (const float*)d_in[3];
  const float* Wv    = (const float*)d_in[4];
  const float* gamma = (const float*)d_in[5];
  const float* beta  = (const float*)d_in[6];
  float* out = (float*)d_out;

  // workspace layout (all offsets 256B-aligned)
  char* ws = (char*)d_ws;
  _Float16* Qh = (_Float16*)(ws);                 // 4 MB
  _Float16* Kh = (_Float16*)(ws + 4194304);       // 4 MB
  float*    V  = (float*)   (ws + 8388608);       // 8 MB
  _Float16* Wt = (_Float16*)(ws + 16777216);      // 96 KB

  wconv<<<192, 256, 0, stream>>>(Wq, Wk, Wv, Wt);
  proj_kernel<<<(B_ * A_) / 16, 256, 0, stream>>>(x, Wt, Qh, Kh, V);
  attn_kernel<<<dim3(A_ / 16, B_), 128, 0, stream>>>(Qh, Kh, V, conn, gamma, beta, out);
}
